// GIN_38774964748481
// MI455X (gfx1250) — compile-verified, matrix operand order checked by reading, not run on hardware
//
#include <hip/hip_runtime.h>
#include <hip/hip_bf16.h>

// ---------------------------------------------------------------------------
// GIN (2 layers) + link-prediction head for MI455X (gfx1250, wave32, WMMA).
// N=50000 nodes, E=600000 edges, D=128 features, P=131072 pairs, O=7 outputs.
// GEMMs run on v_wmma_f32_16x16x32_f16; A tiles staged in LDS via the
// Tensor Data Mover (tensor_load_to_lds + s_wait_tensorcnt) when available.
// ---------------------------------------------------------------------------

#define GIN_N 50000
#define GIN_E 600000
#define GIN_D 128
#define GIN_P 131072
#define GIN_O 7
#define GIN_ND (GIN_N * GIN_D)
#define BN_EPS 1e-5f

typedef __attribute__((ext_vector_type(16))) _Float16     v16h;
typedef __attribute__((ext_vector_type(8)))  float        v8f;
typedef __attribute__((ext_vector_type(4)))  unsigned int u32x4;
typedef __attribute__((ext_vector_type(4)))  int          i32x4;
typedef __attribute__((ext_vector_type(8)))  int          i32x8;

union Frag16 { u32x4 q[2]; v16h h; };

#if __has_builtin(__builtin_amdgcn_tensor_load_to_lds) && \
    __has_builtin(__builtin_amdgcn_s_wait_tensorcnt)
#define GIN_HAS_TDM 1
#else
#define GIN_HAS_TDM 0
#endif

// ---------------------------------------------------------------------------
// Utility kernels
// ---------------------------------------------------------------------------

__global__ void zero_f32(float* __restrict__ p, int n) {
    int i = blockIdx.x * blockDim.x + threadIdx.x;
    if (i < n) p[i] = 0.0f;
}

__global__ void f32_to_f16(const float* __restrict__ src, _Float16* __restrict__ dst, int n) {
    int i = blockIdx.x * blockDim.x + threadIdx.x;
    if (i < n) dst[i] = (_Float16)src[i];
}

// agg[dst[e], k] += h[src[e], k]   (one thread per (edge, feature))
__global__ void scatter_add(const float* __restrict__ h,
                            const int* __restrict__ esrc,
                            const int* __restrict__ edst,
                            float* __restrict__ agg, int nEdges) {
    int idx = blockIdx.x * blockDim.x + threadIdx.x;
    int e = idx >> 7;          // /128
    int k = idx & (GIN_D - 1);
    if (e >= nEdges) return;
    float v = h[(long)esrc[e] * GIN_D + k];
    atomicAdd(&agg[(long)edst[e] * GIN_D + k], v);
}

// a16 = f16((1+eps)*h + agg)
__global__ void combine_eps(const float* __restrict__ h,
                            const float* __restrict__ agg,
                            const float* __restrict__ epsp,
                            _Float16* __restrict__ a16, int n) {
    int i = blockIdx.x * blockDim.x + threadIdx.x;
    if (i >= n) return;
    float e = 1.0f + epsp[0];
    a16[i] = (_Float16)(e * h[i] + agg[i]);
}

// ---------------------------------------------------------------------------
// WMMA GEMM:  C[nrows x 128] = relu( A[nrows x 128] @ W^T + bias )
//   A row-major f16, W row-major f16 [out][in] (so B[k][n] = W[n][k]),
//   bias f32, C f32.  Block = 256 threads = 8 waves; block covers a 16-row
//   tile; wave w computes the 16x16 output tile at columns [16w, 16w+16).
// ---------------------------------------------------------------------------
__global__ void __launch_bounds__(256)
gemm_relu_wmma(const _Float16* __restrict__ A,
               const _Float16* __restrict__ W,
               const float* __restrict__ bias,
               float* __restrict__ Cout, int nrows) {
    __shared__ _Float16 aTile[16 * GIN_D];   // 4 KB

    const int tid  = threadIdx.x;
    const int row0 = blockIdx.x * 16;
    const int rem  = nrows - row0;           // rows remaining (>=1)

#if GIN_HAS_TDM
    // ---- Tensor Data Mover: DMA the 16x128 f16 tile global -> LDS. ----
    // D# per CDNA5 ISA ch.8: 2-D tensor, data_size=2B, zero-fill OOB rows.
    if (tid < 32) {                           // wave 0 issues + waits
        unsigned long long ga =
            (unsigned long long)(const void*)(A + (long)row0 * GIN_D);
        unsigned int ldsa = (unsigned int)(unsigned long long)(__SIZE_TYPE__)aTile;
        unsigned int tileRows = rem < 16 ? (unsigned int)rem : 16u;

        u32x4 g0;
        g0.x = 1u;                            // count=1 (valid), user mode
        g0.y = ldsa;                          // lds_addr  [63:32]
        g0.z = (unsigned int)(ga & 0xFFFFFFFFu);              // global_addr lo
        g0.w = (unsigned int)((ga >> 32) & 0x01FFFFFFu)       // global_addr hi
             | (2u << 30);                    // type=2 ("image")

        i32x8 g1;
        g1[0] = (1 << 16);                    // data_size=1 -> 2 bytes/elt
        g1[1] = (GIN_D << 16);                // tensor_dim0 = 128 (lo16)
        g1[2] = (int)(((unsigned)rem & 0xFFFFu) << 16);   // tensor_dim1 lo16
        g1[3] = (GIN_D << 16) | (int)(((unsigned)rem >> 16) & 0xFFFFu); // tile_dim0=128 | tensor_dim1 hi16
        g1[4] = (int)tileRows;                // tile_dim1 (rows), tile_dim2=0
        g1[5] = GIN_D;                        // tensor_dim0_stride = 128 elts
        g1[6] = 0;                            // stride hi / dim1_stride lo
        g1[7] = 0;                            // dim1_stride hi

        i32x4 z4 = {0, 0, 0, 0};              // groups 2/3: zero (2-D tensor)
#if defined(__clang_major__) && (__clang_major__ >= 23)
        i32x8 z8 = {0, 0, 0, 0, 0, 0, 0, 0};
        __builtin_amdgcn_tensor_load_to_lds(g0, g1, z4, z4, z8, 0);
#else
        __builtin_amdgcn_tensor_load_to_lds(g0, g1, z4, z4, 0);
#endif
        __builtin_amdgcn_s_wait_tensorcnt(0);
    }
#else
    // ---- Fallback: cooperative vector load of the tile (1024 dwords). ----
    {
        const unsigned int* A32 = reinterpret_cast<const unsigned int*>(A);
        unsigned int* S32 = reinterpret_cast<unsigned int*>(aTile);
#pragma unroll
        for (int t = 0; t < 4; ++t) {
            int i = tid + t * 256;           // 0..1023
            int r = i >> 6;                  // row in tile
            int c = i & 63;                  // dword in row
            unsigned int v = 0u;
            if (r < rem) v = A32[(long)(row0 + r) * 64 + c];
            S32[i] = v;
        }
    }
#endif
    __syncthreads();

    const int wave = tid >> 5;               // 0..7 -> 16-col tile
    const int lane = tid & 31;               // wave32
    const int col0 = wave * 16;
    const int half = (lane >> 4) & 1;        // lanes 16-31 hold K(+8) slices
    const int l16  = lane & 15;

    const float bv = bias[col0 + l16];
    v8f acc = { bv, bv, bv, bv, bv, bv, bv, bv };

    const _Float16* __restrict__ arow = aTile + l16 * GIN_D;            // A row l16
    const _Float16* __restrict__ wrow = W + (long)(col0 + l16) * GIN_D; // B col l16

#pragma unroll
    for (int kc = 0; kc < 4; ++kc) {         // K = 128 in chunks of 32
        const int kb = kc * 32 + half * 8;
        Frag16 af, bf;
        // ISA A-frag layout (16-bit, 16x32): half 0 -> K {kb..kb+7, kb+16..kb+23}
        af.q[0] = *reinterpret_cast<const u32x4*>(arow + kb);
        af.q[1] = *reinterpret_cast<const u32x4*>(arow + kb + 16);
        bf.q[0] = *reinterpret_cast<const u32x4*>(wrow + kb);
        bf.q[1] = *reinterpret_cast<const u32x4*>(wrow + kb + 16);
        acc = __builtin_amdgcn_wmma_f32_16x16x32_f16(
            /*neg_a=*/false, af.h, /*neg_b=*/false, bf.h,
            /*c_mod=*/(short)0, acc, /*reuse_a=*/false, /*reuse_b=*/false);
    }

    // C/D layout: VGPR j -> row j (lanes 0-15) / row j+8 (lanes 16-31), col l16.
    float* __restrict__ cbase = Cout + (long)row0 * GIN_D + col0 + l16;
    if (rem >= 16) {                          // uniform fast path: no predication
#pragma unroll
        for (int j = 0; j < 8; ++j) {
            cbase[(long)(j + half * 8) * GIN_D] = fmaxf(acc[j], 0.0f);
        }
    } else {
#pragma unroll
        for (int j = 0; j < 8; ++j) {
            int r = j + half * 8;
            if (r < rem) cbase[(long)r * GIN_D] = fmaxf(acc[j], 0.0f);
        }
    }
}

// ---------------------------------------------------------------------------
// BatchNorm over axis 0 (per-feature across all N rows)
// ---------------------------------------------------------------------------

// stats[0:128] += column sums, stats[128:256] += column sums of squares
__global__ void bn_stats(const float* __restrict__ t, float* __restrict__ stats,
                         int nrows) {
    __shared__ float ssum[GIN_D];
    __shared__ float ssq[GIN_D];
    int tid = threadIdx.x;                   // 256 threads = 2 rows x 128 feats
    if (tid < GIN_D) { ssum[tid] = 0.0f; ssq[tid] = 0.0f; }
    __syncthreads();

    int k = tid & (GIN_D - 1);
    float s = 0.0f, q = 0.0f;
    for (int r = blockIdx.x * 2 + (tid >> 7); r < nrows; r += gridDim.x * 2) {
        float v = t[(long)r * GIN_D + k];
        s += v;
        q += v * v;
    }
    atomicAdd(&ssum[k], s);                  // LDS f32 atomic (ds_add_f32)
    atomicAdd(&ssq[k], q);
    __syncthreads();
    if (tid < GIN_D) {
        atomicAdd(&stats[tid], ssum[tid]);
        atomicAdd(&stats[GIN_D + tid], ssq[tid]);
    }
}

// scsh[0:128] = gamma*rsqrt(var+eps), scsh[128:256] = beta - mu*scale
__global__ void bn_finalize(const float* __restrict__ stats,
                            const float* __restrict__ gamma,
                            const float* __restrict__ beta,
                            float* __restrict__ scsh, int nrows) {
    int k = threadIdx.x;                     // 128 threads
    float inv = 1.0f / (float)nrows;
    float mu  = stats[k] * inv;
    float var = stats[GIN_D + k] * inv - mu * mu;
    float sc  = gamma[k] * rsqrtf(var + BN_EPS);
    scsh[k] = sc;
    scsh[GIN_D + k] = beta[k] - mu * sc;
}

// hout = t*scale + shift (f32), h16 = f16(hout)
__global__ void bn_apply(const float* __restrict__ t,
                         const float* __restrict__ scsh,
                         float* __restrict__ hout,
                         _Float16* __restrict__ h16, int n) {
    int i = blockIdx.x * blockDim.x + threadIdx.x;
    if (i >= n) return;
    int k = i & (GIN_D - 1);
    float v = t[i] * scsh[k] + scsh[GIN_D + k];
    hout[i] = v;
    h16[i]  = (_Float16)v;
}

// ---------------------------------------------------------------------------
// Pair head: out[p, o] = sum_k h[p1,k]*h[p2,k]*Wfc[o,k] + bfc[o]
// One wave32 per pair; cross-lane reduce with shfl_xor.
// ---------------------------------------------------------------------------
__global__ void __launch_bounds__(256)
pair_fc(const float* __restrict__ h, const int* __restrict__ pairs,
        const float* __restrict__ Wfc, const float* __restrict__ bfc,
        float* __restrict__ out, int nPairs) {
    int warp = (blockIdx.x * blockDim.x + threadIdx.x) >> 5;
    int lane = threadIdx.x & 31;
    if (warp >= nPairs) return;
    long p1 = pairs[2 * warp];
    long p2 = pairs[2 * warp + 1];

    float acc[GIN_O];
#pragma unroll
    for (int o = 0; o < GIN_O; ++o) acc[o] = 0.0f;

#pragma unroll
    for (int kk = 0; kk < GIN_D / 32; ++kk) {
        int k = lane + kk * 32;
        float v = h[p1 * GIN_D + k] * h[p2 * GIN_D + k];
#pragma unroll
        for (int o = 0; o < GIN_O; ++o) acc[o] += v * Wfc[o * GIN_D + k];
    }
#pragma unroll
    for (int o = 0; o < GIN_O; ++o) {
        float a = acc[o];
#pragma unroll
        for (int off = 16; off > 0; off >>= 1) a += __shfl_xor(a, off, 32);
        if (lane == 0) out[(long)warp * GIN_O + o] = a + bfc[o];
    }
}

// ---------------------------------------------------------------------------
// Host-side orchestration
// ---------------------------------------------------------------------------
extern "C" void kernel_launch(void* const* d_in, const int* in_sizes, int n_in,
                              void* d_out, int out_size, void* d_ws, size_t ws_size,
                              hipStream_t stream) {
    // Inputs in setup_inputs() order
    const float* x    = (const float*)d_in[0];
    const float* W1a  = (const float*)d_in[1];
    const float* b1a  = (const float*)d_in[2];
    const float* W2a  = (const float*)d_in[3];
    const float* b2a  = (const float*)d_in[4];
    const float* g0   = (const float*)d_in[5];
    const float* bt0  = (const float*)d_in[6];
    const float* eps0 = (const float*)d_in[7];
    const float* W1b  = (const float*)d_in[8];
    const float* b1b  = (const float*)d_in[9];
    const float* W2b  = (const float*)d_in[10];
    const float* b2b  = (const float*)d_in[11];
    const float* g1   = (const float*)d_in[12];
    const float* bt1  = (const float*)d_in[13];
    const float* eps1 = (const float*)d_in[14];
    const float* Wl   = (const float*)d_in[15];
    const float* bl   = (const float*)d_in[16];
    const float* Wfc  = (const float*)d_in[17];
    const float* bfc  = (const float*)d_in[18];
    const int*   esrc = (const int*)d_in[19];
    const int*   edst = (const int*)d_in[20];
    const int*   prs  = (const int*)d_in[21];
    float* out = (float*)d_out;

    // Workspace layout (all segments 16B-aligned)
    float* ws   = (float*)d_ws;
    float* hcur = ws;                    // N*D f32
    float* agg  = hcur + GIN_ND;         // N*D f32
    float* t1   = agg + GIN_ND;          // N*D f32
    float* stats = t1 + GIN_ND;          // 256 f32
    float* scsh  = stats + 2 * GIN_D;    // 256 f32
    _Float16* a16 = (_Float16*)(scsh + 2 * GIN_D);  // N*D f16
    _Float16* h16 = a16 + GIN_ND;                   // N*D f16
    _Float16* wh0 = h16 + GIN_ND;                   // 5 x 128*128 f16
    _Float16* wh1 = wh0 + GIN_D * GIN_D;
    _Float16* wh2 = wh1 + GIN_D * GIN_D;
    _Float16* wh3 = wh2 + GIN_D * GIN_D;
    _Float16* wh4 = wh3 + GIN_D * GIN_D;

    const int BLK = 256;
    const int gridND   = (GIN_ND + BLK - 1) / BLK;       // elementwise over N*D
    const int gridEdge = (GIN_E * GIN_D + BLK - 1) / BLK;
    const int gridGemm = (GIN_N + 15) / 16;              // 3125
    const int gridW    = (GIN_D * GIN_D + BLK - 1) / BLK;
    const int gridPair = (GIN_P * 32 + BLK - 1) / BLK;

    // Weights -> f16 (once per launch; deterministic)
    f32_to_f16<<<gridW, BLK, 0, stream>>>(W1a, wh0, GIN_D * GIN_D);
    f32_to_f16<<<gridW, BLK, 0, stream>>>(W2a, wh1, GIN_D * GIN_D);
    f32_to_f16<<<gridW, BLK, 0, stream>>>(W1b, wh2, GIN_D * GIN_D);
    f32_to_f16<<<gridW, BLK, 0, stream>>>(W2b, wh3, GIN_D * GIN_D);
    f32_to_f16<<<gridW, BLK, 0, stream>>>(Wl,  wh4, GIN_D * GIN_D);

    // ---------------- GIN layer 1 ----------------
    zero_f32<<<gridND, BLK, 0, stream>>>(agg, GIN_ND);
    scatter_add<<<gridEdge, BLK, 0, stream>>>(x, esrc, edst, agg, GIN_E);
    combine_eps<<<gridND, BLK, 0, stream>>>(x, agg, eps0, a16, GIN_ND);
    gemm_relu_wmma<<<gridGemm, BLK, 0, stream>>>(a16, wh0, b1a, t1, GIN_N);
    f32_to_f16<<<gridND, BLK, 0, stream>>>(t1, a16, GIN_ND);
    gemm_relu_wmma<<<gridGemm, BLK, 0, stream>>>(a16, wh1, b2a, t1, GIN_N);
    zero_f32<<<1, BLK, 0, stream>>>(stats, 2 * GIN_D);
    bn_stats<<<512, BLK, 0, stream>>>(t1, stats, GIN_N);
    bn_finalize<<<1, GIN_D, 0, stream>>>(stats, g0, bt0, scsh, GIN_N);
    bn_apply<<<gridND, BLK, 0, stream>>>(t1, scsh, hcur, h16, GIN_ND);

    // ---------------- GIN layer 2 ----------------
    zero_f32<<<gridND, BLK, 0, stream>>>(agg, GIN_ND);
    scatter_add<<<gridEdge, BLK, 0, stream>>>(hcur, esrc, edst, agg, GIN_E);
    combine_eps<<<gridND, BLK, 0, stream>>>(hcur, agg, eps1, a16, GIN_ND);
    gemm_relu_wmma<<<gridGemm, BLK, 0, stream>>>(a16, wh2, b1b, t1, GIN_N);
    f32_to_f16<<<gridND, BLK, 0, stream>>>(t1, a16, GIN_ND);
    gemm_relu_wmma<<<gridGemm, BLK, 0, stream>>>(a16, wh3, b2b, t1, GIN_N);
    zero_f32<<<1, BLK, 0, stream>>>(stats, 2 * GIN_D);
    bn_stats<<<512, BLK, 0, stream>>>(t1, stats, GIN_N);
    bn_finalize<<<1, GIN_D, 0, stream>>>(stats, g1, bt1, scsh, GIN_N);
    bn_apply<<<gridND, BLK, 0, stream>>>(t1, scsh, hcur, h16, GIN_ND);

    // ---------------- final linear + pair head ----------------
    gemm_relu_wmma<<<gridGemm, BLK, 0, stream>>>(h16, wh4, bl, t1, GIN_N);
    pair_fc<<<gridPair, BLK, 0, stream>>>(t1, prs, Wfc, bfc, out, GIN_P);

    (void)in_sizes; (void)n_in; (void)out_size; (void)ws_size;
}
